// TESwiglu_55164559950495
// MI455X (gfx1250) — matrix operand order repaired and checked by reading, not compiled, not run on hardware
//
#include <hip/hip_runtime.h>
#include <hip/hip_bf16.h>

// Problem constants (from reference)
#define G_    8
#define H_    4096
#define I_    1536
#define D_    (2 * I_)     // 3072
#define MTOT  16384

// Tiling
#define BM    128          // output rows per block
#define BN    64           // output cols per block (gate AND up computed)
#define BK    32           // K per step (one bf16 WMMA deep)
#define PAD   8
#define LDA   (BK + PAD)   // 40 bf16 = 80B row stride -> conflict-free b128 LDS loads

typedef __bf16 bf16;
typedef __bf16 v16bf __attribute__((ext_vector_type(16)));
typedef float  v8f   __attribute__((ext_vector_type(8)));

union Frag  { v16bf v; uint4 q[2]; };
union Pack4 { bf16 h[4]; uint2 u; };

__global__ __launch_bounds__(256)
void swiglu_gemm(const float* __restrict__ x,
                 const float* __restrict__ w,
                 float* __restrict__ out)
{
    __shared__ __align__(16) bf16 As[2][BM][LDA];   // x tile (128 x 32)
    __shared__ __align__(16) bf16 Bs[2][BM][LDA];   // w tile: rows 0..63 gate, 64..127 up

    const int tid  = threadIdx.x;
    const int lane = tid & 31;
    const int wid  = tid >> 5;
    const int wm   = wid & 1;    // 0..1 : which 64-row half
    const int wn   = wid >> 1;   // 0..3 : which 16-col slice

    const int row0 = blockIdx.y * BM;        // global row (groups are contiguous row blocks)
    const int g    = row0 / (MTOT / G_);     // group index
    const int n0   = blockIdx.x * BN;        // output column base (0..1535)

    const float* xg = x + (size_t)row0 * H_;
    const float* wg = w + (size_t)g * D_ * H_;

    v8f accG[4], accU[4];
    const v8f vzero = {0.f,0.f,0.f,0.f,0.f,0.f,0.f,0.f};
#pragma unroll
    for (int s = 0; s < 4; ++s) { accG[s] = vzero; accU[s] = vzero; }

    // ---- cooperative staging: global f32 -> bf16 -> LDS -------------------
    auto stageA = [&](int buf, int k0) {
#pragma unroll
        for (int i = 0; i < 4; ++i) {
            int j  = i * 256 + tid;          // 1024 float4 loads: 128 rows x 8 chunks
            int r  = j >> 3;
            int c4 = (j & 7) * 4;
            float4 f = *reinterpret_cast<const float4*>(xg + (size_t)r * H_ + k0 + c4);
            Pack4 p;
            p.h[0] = (bf16)f.x; p.h[1] = (bf16)f.y; p.h[2] = (bf16)f.z; p.h[3] = (bf16)f.w;
            *reinterpret_cast<uint2*>(&As[buf][r][c4]) = p.u;
        }
    };
    auto stageB = [&](int buf, int k0) {
#pragma unroll
        for (int i = 0; i < 4; ++i) {
            int j  = i * 256 + tid;
            int r  = j >> 3;                 // 0..127: gate rows then up rows
            int c4 = (j & 7) * 4;
            int wrow = (r < BN) ? (n0 + r) : (I_ + n0 + (r - BN));
            float4 f = *reinterpret_cast<const float4*>(wg + (size_t)wrow * H_ + k0 + c4);
            Pack4 p;
            p.h[0] = (bf16)f.x; p.h[1] = (bf16)f.y; p.h[2] = (bf16)f.z; p.h[3] = (bf16)f.w;
            *reinterpret_cast<uint2*>(&Bs[buf][r][c4]) = p.u;
        }
    };

    // ---- one K-step of WMMAs, fragments in contiguous k order (k = 16*half + s) -------------
    auto compute = [&](int buf) {
        const int half = (lane >= 16);
        const int lr   = lane & 15;
        // B fragment (32x16, K x N): lane = column; K 0-15 (lanes 0-15) / 16-31 (lanes 16-31)
        Frag bg, bu;
        {
            int koff = half ? 16 : 0;
            int r = wn * 16 + lr;
            bg.q[0] = *reinterpret_cast<const uint4*>(&Bs[buf][r][koff]);
            bg.q[1] = *reinterpret_cast<const uint4*>(&Bs[buf][r][koff + 8]);
            bu.q[0] = *reinterpret_cast<const uint4*>(&Bs[buf][r + BN][koff]);
            bu.q[1] = *reinterpret_cast<const uint4*>(&Bs[buf][r + BN][koff + 8]);
        }
#pragma unroll
        for (int s = 0; s < 4; ++s) {
            // A fragment (16x32, M x K): lanes 0-15 hold K 0-15, lanes 16-31 K 16-31
            Frag a;
            int koff = half ? 16 : 0;
            int r = wm * 64 + s * 16 + lr;
            a.q[0] = *reinterpret_cast<const uint4*>(&As[buf][r][koff]);
            a.q[1] = *reinterpret_cast<const uint4*>(&As[buf][r][koff + 8]);
            accG[s] = __builtin_amdgcn_wmma_f32_16x16x32_bf16(
                false, a.v, false, bg.v, (short)0, accG[s], false, false);
            accU[s] = __builtin_amdgcn_wmma_f32_16x16x32_bf16(
                false, a.v, false, bu.v, (short)0, accU[s], false, false);
        }
    };

    // ---- double-buffered main loop ----------------------------------------
    stageA(0, 0);
    stageB(0, 0);
    __syncthreads();

    const int KSTEPS = H_ / BK;   // 128
    for (int kt = 0; kt < KSTEPS; ++kt) {
        int buf = kt & 1;
        if (kt + 1 < KSTEPS) {
            stageA(buf ^ 1, (kt + 1) * BK);
            stageB(buf ^ 1, (kt + 1) * BK);
        }
        compute(buf);
        __syncthreads();
    }

    // ---- fused SwiGLU epilogue; C/D layout: (VGPR j, lane l) -> M=j+8*(l>=16), N=l%16
    const int half = (lane >= 16);
    const int lr   = lane & 15;
#pragma unroll
    for (int s = 0; s < 4; ++s) {
#pragma unroll
        for (int j = 0; j < 8; ++j) {
            float gk  = accG[s][j];
            float uk  = accU[s][j];
            float sig = 1.0f / (1.0f + __expf(-gk));
            float o   = gk * sig * uk;
            int rrow  = row0 + wm * 64 + s * 16 + j + (half ? 8 : 0);
            int col   = n0 + wn * 16 + lr;
            out[(size_t)rrow * I_ + col] = o;
        }
    }
}

extern "C" void kernel_launch(void* const* d_in, const int* in_sizes, int n_in,
                              void* d_out, int out_size, void* d_ws, size_t ws_size,
                              hipStream_t stream) {
    const float* x = (const float*)d_in[0];
    const float* w = (const float*)d_in[1];
    // d_in[2] = m_splits scalar (compile-time constant here)
    float* out = (float*)d_out;

    dim3 grid(I_ / BN, MTOT / BM);   // (24, 128)
    dim3 block(256);                 // 8 wave32 waves
    hipLaunchKernelGGL(swiglu_gemm, grid, block, 0, stream, x, w, out);
}
